// Attention_31001073943086
// MI455X (gfx1250) — matrix operand order repaired and checked by reading, not compiled, not run on hardware
//
#include <hip/hip_runtime.h>

// ---------------------------------------------------------------------------
// Problem constants (from the reference): B=2, T=2048, D=2048, H=16, KV=4, HD=128
// ---------------------------------------------------------------------------
#define CB   2
#define CT   2048
#define CD   2048
#define CH   16
#define CKV  4
#define CHD  128

typedef __attribute__((ext_vector_type(16))) __bf16        v16bf;
typedef __attribute__((ext_vector_type(8)))  float         v8f;
typedef __attribute__((ext_vector_type(8)))  unsigned int  v8u;

__device__ __forceinline__ unsigned short f2bf(float f) {
  unsigned int u = __builtin_bit_cast(unsigned int, f);
  u += 0x7FFFu + ((u >> 16) & 1u);           // round-to-nearest-even
  return (unsigned short)(u >> 16);
}

__device__ __forceinline__ v8f wmma_bf16(v16bf a, v16bf b, v8f c) {
  // D(f32 16x16) = A(16x32 bf16) * B(32x16 bf16) + C
  return __builtin_amdgcn_wmma_f32_16x16x32_bf16(
      /*neg_a=*/false, a, /*neg_b=*/false, b,
      /*c_mod=*/(short)0, c, /*reuse_a=*/false, /*reuse_b=*/false);
}

// ---- CDNA5 async global->LDS copy (ASYNCcnt-tracked, no VGPR round-trip) ----
// dst: LDS byte offset (low 32 bits of the flat shared pointer = addrspace(3)
// offset); src: 64-bit global address. Copies 16 bytes per active lane.
__device__ __forceinline__ void async_g2l_b128(void* lds_dst, const void* gsrc) {
  unsigned int       ldso = (unsigned int)(unsigned long long)lds_dst;
  unsigned long long ga   = (unsigned long long)gsrc;
  asm volatile("global_load_async_to_lds_b128 %0, %1, off"
               :: "v"(ldso), "v"(ga)
               : "memory");
}
__device__ __forceinline__ void wait_asynccnt0() {
  asm volatile("s_wait_asynccnt 0x0" ::: "memory");
}

// A-matrix fragment (16x32 bf16, ISA 7.12.2): lane m=lane&15;
//   VGPR v holds K pair at (v>>2)*16 + (lane>>4)*8 + (v&3)*2
__device__ __forceinline__ v16bf load_frag_a(const unsigned short* base,
                                             int rowbase, int ld, int k0, int lane) {
  const int half = lane >> 4, m = lane & 15;
  const unsigned short* p = base + (rowbase + m) * ld + k0;
  v8u u;
#pragma unroll
  for (int v = 0; v < 8; ++v) {
    const int kb = ((v >> 2) << 4) + (half << 3) + ((v & 3) << 1);
    u[v] = *(const unsigned int*)(p + kb);
  }
  return __builtin_bit_cast(v16bf, u);
}

// B-matrix fragment (32x16 bf16) from K-contiguous [N][K] storage:
//   lane n=lane&15; VGPR v holds K pair at (v>>2)*16 + (lane>>4)*8 + (v&3)*2
__device__ __forceinline__ v16bf load_frag_b(const unsigned short* base,
                                             int colbase, int ld, int k0, int lane) {
  const int half = lane >> 4, n = lane & 15;
  const unsigned short* p = base + (colbase + n) * ld + k0;
  v8u u;
#pragma unroll
  for (int v = 0; v < 8; ++v) {
    const int kb = ((v >> 2) << 4) + (half << 3) + ((v & 3) << 1);
    u[v] = *(const unsigned int*)(p + kb);
  }
  return __builtin_bit_cast(v16bf, u);
}

// ---------------------------------------------------------------------------
// Projection GEMM: 64x128 tile / block, K-loop in steps of 32.
// MODE 0: Q proj  (rms-norm + rope, heads=16) -> bf16 [B,H,T,HD]
// MODE 1: K proj  (rms-norm + rope, heads=4)  -> bf16 [B,KV,T,HD]
// MODE 2: V proj  (plain, heads=4)            -> bf16 [B,KV,T,HD]
// MODE 3: out proj (bf16 activations x fp32 W)-> fp32 [B*T, D]
// ---------------------------------------------------------------------------
template <int MODE, int NTOT, int HOUT>
__global__ __launch_bounds__(256) void gemm_proj_kernel(
    const float* __restrict__ Xf, const unsigned short* __restrict__ Xbf,
    const float* __restrict__ W,
    const float* __restrict__ nscale,
    const float* __restrict__ cosT, const float* __restrict__ sinT,
    unsigned short* __restrict__ Obf, float* __restrict__ Of) {
  constexpr int KTOT = 2048;   // D for qkv, H*HD for out-proj
  const int tid  = threadIdx.x;
  const int lane = tid & 31;
  const int wid  = tid >> 5;
  const int n0   = blockIdx.x * 128;
  const int m0   = blockIdx.y * 64;

  __shared__ unsigned short xs[64][32];     // A tile, bf16, [M][K]
  __shared__ unsigned short wt[128][32];    // B tile, bf16, [N][K] (K-contiguous)
  __shared__ float          otile[64][128];
  __shared__ float          red[64][4];

  const int wrow = wid >> 1;   // 0..3 : 16-row strip
  const int wcol = wid & 1;    // 0..1 : 64-col strip

  v8f z = {0.f, 0.f, 0.f, 0.f, 0.f, 0.f, 0.f, 0.f};
  v8f acc[4] = {z, z, z, z};

  // per-thread staging coordinates
  const int ar = (tid * 8) >> 5, ac = (tid * 8) & 31;   // 8 contiguous A elems

  for (int k0 = 0; k0 < KTOT; k0 += 32) {
    // stage A (async bf16 copy for MODE 3, fp32->bf16 convert otherwise)
    if constexpr (MODE == 3) {
      async_g2l_b128(&xs[ar][ac], Xbf + (size_t)(m0 + ar) * KTOT + k0 + ac);
    } else {
#pragma unroll
      for (int j = 0; j < 8; ++j)
        xs[ar][ac + j] = f2bf(Xf[(size_t)(m0 + ar) * KTOT + k0 + ac + j]);
    }
    // stage B transposed into [N][K] (global reads coalesced along N)
#pragma unroll
    for (int j = 0; j < 16; ++j) {
      int i = tid * 16 + j, k = i >> 7, n = i & 127;
      wt[n][k] = f2bf(W[(size_t)(k0 + k) * NTOT + n0 + n]);
    }
    // prefetch next K-step tiles into the cache hierarchy
    if (k0 + 32 < KTOT) {
      if constexpr (MODE != 3)
        __builtin_prefetch(Xf + (size_t)(m0 + ar) * KTOT + k0 + 32 + ac, 0, 1);
      __builtin_prefetch(W + (size_t)(k0 + 32 + ((tid * 16) >> 7)) * NTOT + n0 +
                             ((tid * 16) & 127), 0, 1);
    }
    if constexpr (MODE == 3) wait_asynccnt0();
    __syncthreads();

    v16bf a = load_frag_a(&xs[0][0], wrow * 16, 32, 0, lane);
#pragma unroll
    for (int nt = 0; nt < 4; ++nt) {
      v16bf b = load_frag_b(&wt[0][0], wcol * 64 + nt * 16, 32, 0, lane);
      acc[nt] = wmma_bf16(a, b, acc[nt]);
    }
    __syncthreads();
  }

  // dump accumulators to LDS (C layout: VGPR v -> row v + 8*(lane>>4), col lane&15)
  {
    const int half = lane >> 4, mn = lane & 15;
#pragma unroll
    for (int nt = 0; nt < 4; ++nt)
#pragma unroll
      for (int v = 0; v < 8; ++v)
        otile[wrow * 16 + v + 8 * half][wcol * 64 + nt * 16 + mn] = acc[nt][v];
  }
  __syncthreads();

  const int row = tid >> 2, seg = tid & 3;
  const int gr  = m0 + row;                 // global row = b*T + t
  if constexpr (MODE == 3) {
#pragma unroll 4
    for (int c = seg * 32; c < seg * 32 + 32; ++c)
      Of[(size_t)gr * NTOT + n0 + c] = otile[row][c];
  } else {
    const int t = gr & (CT - 1);
    const int b = gr >> 11;                 // / 2048
    const int h = blockIdx.x;               // 128-wide tile == one head
    unsigned short* outp = Obf + (((size_t)(b * HOUT + h)) * CT + t) * CHD;
    if constexpr (MODE == 2) {
#pragma unroll 4
      for (int c = seg * 32; c < seg * 32 + 32; ++c)
        outp[c] = f2bf(otile[row][c]);
    } else {
      // RMS norm over the head dim (all 128 cols live in this tile)
      float ss = 0.f;
#pragma unroll 4
      for (int c = seg * 32; c < seg * 32 + 32; ++c) {
        float vv = otile[row][c];
        ss += vv * vv;
      }
      red[row][seg] = ss;
      __syncthreads();
      float var = (red[row][0] + red[row][1] + red[row][2] + red[row][3]) * (1.f / 128.f);
      float rs  = rsqrtf(var + 1e-6f);
      // RoPE (cos/sin tables already repeated pairwise: cos[2i]==cos[2i+1])
#pragma unroll 4
      for (int c = seg * 32; c < seg * 32 + 32; c += 2) {
        float v0 = otile[row][c] * rs * nscale[c];
        float v1 = otile[row][c + 1] * rs * nscale[c + 1];
        float c0 = cosT[t * CHD + c], c1 = cosT[t * CHD + c + 1];
        float s0 = sinT[t * CHD + c], s1 = sinT[t * CHD + c + 1];
        outp[c]     = f2bf(v0 * c0 - v1 * s0);
        outp[c + 1] = f2bf(v1 * c1 + v0 * s1);
      }
    }
  }
}

// ---------------------------------------------------------------------------
// Flash attention: one (b, h, 64-row q-tile) per block; causal k-tile loop.
// ---------------------------------------------------------------------------
__global__ __launch_bounds__(256) void attn_fa_kernel(
    const unsigned short* __restrict__ Q,   // [B,H,T,HD] bf16
    const unsigned short* __restrict__ K,   // [B,KV,T,HD] bf16
    const unsigned short* __restrict__ V,   // [B,KV,T,HD] bf16
    unsigned short* __restrict__ AO) {      // [B,T,H*HD] bf16
  const int tid  = threadIdx.x;
  const int lane = tid & 31;
  const int wid  = tid >> 5;
  const int half = lane >> 4, mn = lane & 15;

  const int qi  = blockIdx.x & 31;           // T/64 tiles
  const int h   = (blockIdx.x >> 5) & (CH - 1);
  const int b   = blockIdx.x >> 9;
  const int q0  = qi * 64;
  const int kvh = h >> 2;                    // h / (H/KV)

  __shared__ unsigned short qs[64][128];     // [q][d] bf16
  __shared__ unsigned short ks[64][128];     // [k][d] bf16 (B for QK^T: N=k, K=d)
  __shared__ unsigned short vsT[128][64];    // [d][k] bf16 (B for PV:  N=d, K=k)
  __shared__ float          st[64][64];      // S tile fp32
  __shared__ unsigned short ps[64][64];      // P tile bf16
  __shared__ float          m_i[64], l_i[64], alf[64];
  __shared__ float          red[64][4];

  const int crow = tid >> 2, cseg = tid & 3; // copy coords: 64B / thread / row
  const size_t qbase = (((size_t)(b * CH + h)) * CT + q0) * CHD;
  // async copy of the whole Q tile (4 x 16B per thread), no VGPR round-trip
#pragma unroll
  for (int j = 0; j < 4; ++j)
    async_g2l_b128(&qs[crow][cseg * 32 + j * 8],
                   Q + qbase + crow * CHD + cseg * 32 + j * 8);
  if (tid < 64) { m_i[tid] = -1e30f; l_i[tid] = 0.f; }
  wait_asynccnt0();

  v8f z = {0.f, 0.f, 0.f, 0.f, 0.f, 0.f, 0.f, 0.f};
  v8f oacc[4] = {z, z, z, z};
  const int orow = wid >> 1;       // O tile-row (16 q rows)
  const int ocol = (wid & 1) * 4;  // first of 4 O tile-cols (16 d each)

  const int nkt = qi + 1;          // causal
  for (int kt = 0; kt < nkt; ++kt) {
    const int k0 = kt * 64;
    __syncthreads();               // prev iter done with ks/vsT/ps (and qs ready)
    const size_t kb = (((size_t)(b * CKV + kvh)) * CT + k0) * CHD;
    // K tile: straight bf16 copy -> async to LDS
#pragma unroll
    for (int j = 0; j < 4; ++j)
      async_g2l_b128(&ks[crow][cseg * 32 + j * 8],
                     K + kb + crow * CHD + cseg * 32 + j * 8);
    // V tile: transposed into [d][t] for PV B-fragments (VGPR path)
#pragma unroll 8
    for (int j = 0; j < 32; ++j) {
      int i = tid + j * 256, t = i >> 7, d = i & 127;
      vsT[d][t] = V[kb + i];
    }
    wait_asynccnt0();
    __syncthreads();

    // ---- S = Q K^T : 4x4 grid of 16x16 tiles, 2 tiles per wave ----
#pragma unroll
    for (int s = 0; s < 2; ++s) {
      const int ti = wid * 2 + s, trow = ti >> 2, tcol = ti & 3;
      v8f sa = z;
#pragma unroll
      for (int kc = 0; kc < 128; kc += 32) {
        v16bf a  = load_frag_a(&qs[0][0], trow * 16, 128, kc, lane);
        v16bf bb = load_frag_b(&ks[0][0], tcol * 16, 128, kc, lane);
        sa = wmma_bf16(a, bb, sa);
      }
#pragma unroll
      for (int v = 0; v < 8; ++v) {
        const int rl = trow * 16 + v + 8 * half;
        const int cl = tcol * 16 + mn;
        float sv = sa[v] * 0.08838834764831845f;   // 1/sqrt(128)
        sv = 50.f * tanhf(sv * 0.02f);             // soft cap 50
        if (k0 + cl > q0 + rl) sv = -1e30f;        // causal mask
        st[rl][cl] = sv;
      }
    }
    __syncthreads();

    // ---- online softmax: 4 threads / row, 16 cols each ----
    const int row = crow, seg = cseg;
    float pm = -1e30f;
#pragma unroll 4
    for (int c = seg * 16; c < seg * 16 + 16; ++c) pm = fmaxf(pm, st[row][c]);
    red[row][seg] = pm;
    __syncthreads();
    const float m_old = m_i[row];
    const float m_new = fmaxf(fmaxf(fmaxf(red[row][0], red[row][1]),
                                    fmaxf(red[row][2], red[row][3])), m_old);
    const float a_ = __expf(m_old - m_new);
    float psum = 0.f;
#pragma unroll 4
    for (int c = seg * 16; c < seg * 16 + 16; ++c) {
      float p = __expf(st[row][c] - m_new);
      ps[row][c] = f2bf(p);
      psum += p;
    }
    __syncthreads();               // everyone done reading red (max)
    red[row][seg] = psum;
    if (seg == 0) alf[row] = a_;
    __syncthreads();               // ps/alf/red(sum) visible
    if (seg == 0) {
      l_i[row] = l_i[row] * a_ + red[row][0] + red[row][1] + red[row][2] + red[row][3];
      m_i[row] = m_new;
    }

    // ---- O = O*alpha + P V ----
    float av[8];
#pragma unroll
    for (int v = 0; v < 8; ++v) av[v] = alf[orow * 16 + v + 8 * half];
#pragma unroll
    for (int ot = 0; ot < 4; ++ot)
#pragma unroll
      for (int v = 0; v < 8; ++v) oacc[ot][v] *= av[v];
#pragma unroll
    for (int kc = 0; kc < 64; kc += 32) {
      v16bf a = load_frag_a(&ps[0][0], orow * 16, 64, kc, lane);
#pragma unroll
      for (int ot = 0; ot < 4; ++ot) {
        v16bf bb = load_frag_b(&vsT[0][0], (ocol + ot) * 16, 64, kc, lane);
        oacc[ot] = wmma_bf16(a, bb, oacc[ot]);
      }
    }
  }
  __syncthreads();                 // l_i final

  float linv[8];
#pragma unroll
  for (int v = 0; v < 8; ++v) linv[v] = 1.f / l_i[orow * 16 + v + 8 * half];
#pragma unroll
  for (int ot = 0; ot < 4; ++ot)
#pragma unroll
    for (int v = 0; v < 8; ++v) {
      const int rl = orow * 16 + v + 8 * half;
      const int t  = q0 + rl;
      const int cl = (ocol + ot) * 16 + mn;
      AO[((size_t)(b * CT + t)) * (CH * CHD) + h * CHD + cl] =
          f2bf(oacc[ot][v] * linv[v]);
    }
}

// ---------------------------------------------------------------------------
extern "C" void kernel_launch(void* const* d_in, const int* in_sizes, int n_in,
                              void* d_out, int out_size, void* d_ws, size_t ws_size,
                              hipStream_t stream) {
  (void)in_sizes; (void)n_in; (void)out_size; (void)ws_size;
  const float* x    = (const float*)d_in[0];
  const float* wq   = (const float*)d_in[1];
  const float* wk   = (const float*)d_in[2];
  const float* wv   = (const float*)d_in[3];
  const float* wo   = (const float*)d_in[4];
  const float* qn   = (const float*)d_in[5];
  const float* kn   = (const float*)d_in[6];
  const float* cosT = (const float*)d_in[7];
  const float* sinT = (const float*)d_in[8];
  // d_in[9] = causal mask: computed inline instead.
  float* out = (float*)d_out;

  // workspace carve (bf16 intermediates): q 16MB, k 4MB, v 4MB, ao 16MB
  unsigned short* qb = (unsigned short*)d_ws;
  unsigned short* kb = qb + (size_t)CB * CH  * CT * CHD;
  unsigned short* vb = kb + (size_t)CB * CKV * CT * CHD;
  unsigned short* ao = vb + (size_t)CB * CKV * CT * CHD;

  dim3 blk(256);
  // projections (Mtot = B*T = 4096 -> 64 row tiles)
  gemm_proj_kernel<0, 2048, 16><<<dim3(16, 64), blk, 0, stream>>>(
      x, nullptr, wq, qn, cosT, sinT, qb, nullptr);
  gemm_proj_kernel<1, 512, 4><<<dim3(4, 64), blk, 0, stream>>>(
      x, nullptr, wk, kn, cosT, sinT, kb, nullptr);
  gemm_proj_kernel<2, 512, 4><<<dim3(4, 64), blk, 0, stream>>>(
      x, nullptr, wv, nullptr, nullptr, nullptr, vb, nullptr);
  // attention: B*H*(T/64) = 1024 blocks
  attn_fa_kernel<<<dim3(CB * CH * (CT / 64)), blk, 0, stream>>>(qb, kb, vb, ao);
  // output projection -> fp32
  gemm_proj_kernel<3, 2048, 16><<<dim3(16, 64), blk, 0, stream>>>(
      nullptr, ao, wo, nullptr, nullptr, nullptr, nullptr, out);
}
